// TGAN_90031104459216
// MI455X (gfx1250) — matrix operand order repaired and checked by reading, not compiled, not run on hardware
//
#include <hip/hip_runtime.h>
#include <hip/hip_bf16.h>

// ---------------- problem constants ----------------
constexpr int  B_  = 16384;
constexpr int  N_  = 20;
constexpr int  D_  = 128;
constexpr int  H_  = 4;
constexpr int  DM_ = 384;          // 3*D
constexpr int  DK_ = 96;           // DM/H
constexpr long BN_ = (long)B_ * N_;
constexpr float SCALE_ = 0.10206207261596576f;   // 1/sqrt(96)

typedef _Float16 f16;
typedef __attribute__((ext_vector_type(16))) _Float16 v16h;
typedef __attribute__((ext_vector_type(8)))  _Float16 v8h;
typedef __attribute__((ext_vector_type(8)))  float    v8f;

// ---------------- WMMA helpers (CDNA5 16x16x32 f16 -> f32) ----------------
__device__ __forceinline__ v16h cat16(v8h lo, v8h hi) {
  return __builtin_shufflevector(lo, hi, 0,1,2,3,4,5,6,7,8,9,10,11,12,13,14,15);
}
// A fragment: lane r=lane&15 holds row r; half=lane>>4 selects K sub-block.
// element j (0..7) -> K = half*8 + j ; element j (8..15) -> K = 16 + half*8 + (j-8)
__device__ __forceinline__ v16h ldA(const f16* row, int k0, int half) {
  v8h lo = *(const v8h*)(row + k0 + half * 8);
  v8h hi = *(const v8h*)(row + k0 + 16 + half * 8);
  return cat16(lo, hi);
}
// B fragment: lane n=lane&15 holds column n (== row n of W for C=A@W^T);
// element j (0..7) -> K = half*8 + j ; element j (8..15) -> K = 16 + half*8 + (j-8)
__device__ __forceinline__ v16h ldB(const f16* row, int k0, int half) {
  v8h lo = *(const v8h*)(row + k0 + half * 8);
  v8h hi = *(const v8h*)(row + k0 + 16 + half * 8);
  return cat16(lo, hi);
}
__device__ __forceinline__ v8f wmma(v16h a, v16h b, v8f c) {
  return __builtin_amdgcn_wmma_f32_16x16x32_f16(false, a, false, b, (short)0, c, false, false);
}

// ---------------- kernel 0: f32 -> f16 weight convert ----------------
__global__ void cvt_f32_f16(const float* __restrict__ in, f16* __restrict__ out, int n) {
  int i = blockIdx.x * 256 + threadIdx.x;
  if (i < n) out[i] = (f16)in[i];
}

// ---------------- kernel 1a: pack k_in = [seq, seq_e, cos(ts*f+p)] (f16) ----------------
__global__ void pack_kin(const float* __restrict__ seq, const float* __restrict__ seq_e,
                         const float* __restrict__ seq_ts, const float* __restrict__ bf,
                         const float* __restrict__ ph, f16* __restrict__ kin16) {
  long idx = (long)blockIdx.x * 256 + threadIdx.x;
  if (idx >= BN_ * DM_) return;
  long row = idx / DM_;
  int  j   = (int)(idx - row * DM_);
  float v;
  if (j < D_)            v = seq[row * D_ + j];
  else if (j < 2 * D_)   v = seq_e[row * D_ + (j - D_)];
  else                   v = __cosf(seq_ts[row] * bf[j - 2 * D_] + ph[j - 2 * D_]);
  kin16[idx] = (f16)v;
}

// ---------------- kernel 1b: pack q_in (f16 + f32 copy for residual) ----------------
__global__ void pack_qin(const float* __restrict__ src, const float* __restrict__ src_ts,
                         const float* __restrict__ bf, const float* __restrict__ ph,
                         f16* __restrict__ qin16, float* __restrict__ qinf) {
  long idx = (long)blockIdx.x * 256 + threadIdx.x;
  if (idx >= (long)B_ * DM_) return;
  long b = idx / DM_;
  int  j = (int)(idx - b * DM_);
  float v;
  if (j < D_)            v = src[b * D_ + j];
  else if (j < 2 * D_)   v = 0.0f;
  else                   v = __cosf(src_ts[b] * bf[j - 2 * D_] + ph[j - 2 * D_]);
  qin16[idx] = (f16)v;
  qinf[idx]  = v;
}

// ---------------- kernel 1c: src -> f16 (MLP concat tail) ----------------
__global__ void pack_src16(const float* __restrict__ src, f16* __restrict__ src16) {
  long idx = (long)blockIdx.x * 256 + threadIdx.x;
  if (idx < (long)B_ * D_) src16[idx] = (f16)src[idx];
}

// ---------------- kernel 2: C(M x 384) = A(M x 384) @ W(384 x 384)^T, all f16 ----------------
// block = 256 thr = 8 waves; block tile 64(M) x 384(N); wave tile 32 x 96 (2 Mt x 6 Nt)
__global__ __launch_bounds__(256) void gemm_kn384(const f16* __restrict__ A,
                                                  const f16* __restrict__ W,
                                                  f16* __restrict__ C) {
  __shared__ __align__(16) f16 As[64 * 384];
  const int  tid  = threadIdx.x;
  const long row0 = (long)blockIdx.x * 64;
  // cooperative stage of 64 A rows (48 KB) into LDS
  const v8h* srcv = (const v8h*)(A + row0 * DM_);
  v8h*       dstv = (v8h*)As;
#pragma unroll
  for (int c = 0; c < 12; ++c) dstv[tid + c * 256] = srcv[tid + c * 256];
  __syncthreads();

  const int w = tid >> 5, lane = tid & 31;
  const int mg = w >> 2, ng = w & 3;
  const int r = lane & 15, half = lane >> 4;

  const f16* a0 = As + (mg * 32 + r) * DM_;
  const f16* a1 = As + (mg * 32 + 16 + r) * DM_;
  const f16* brow[6];
#pragma unroll
  for (int nt = 0; nt < 6; ++nt) brow[nt] = W + (ng * 96 + nt * 16 + r) * DM_;

  v8f acc[2][6] = {};
  for (int k0 = 0; k0 < DM_; k0 += 32) {
    v16h af0 = ldA(a0, k0, half);
    v16h af1 = ldA(a1, k0, half);
#pragma unroll
    for (int nt = 0; nt < 6; ++nt) {
      v16h bf = ldB(brow[nt], k0, half);
      acc[0][nt] = wmma(af0, bf, acc[0][nt]);
      acc[1][nt] = wmma(af1, bf, acc[1][nt]);
    }
  }
#pragma unroll
  for (int mt = 0; mt < 2; ++mt)
#pragma unroll
    for (int nt = 0; nt < 6; ++nt)
#pragma unroll
      for (int i = 0; i < 8; ++i) {
        long rr = row0 + mg * 32 + mt * 16 + half * 8 + i;
        int  cc = ng * 96 + nt * 16 + r;
        C[rr * DM_ + cc] = (f16)acc[mt][nt][i];
      }
}

// ---------------- kernel 3: per-b attention (scores, softmax, attn_sq, pk) ----------------
// pk[b,h,j] = sum_n p[b,h,n] * k_in[b,n,j]   (V-projection reordering)
__global__ __launch_bounds__(128) void attn_kernel(const f16* __restrict__ Q16,
                                                   const f16* __restrict__ K16,
                                                   const f16* __restrict__ kin16,
                                                   const unsigned char* __restrict__ mask,
                                                   float* __restrict__ attn_sq,
                                                   f16* __restrict__ pk16) {
  const int b = blockIdx.x, tid = threadIdx.x;
  __shared__ float sQ[DM_];
  __shared__ float sS[H_][N_];
  __shared__ float sP[H_][N_];
  for (int j = tid; j < DM_; j += 128) sQ[j] = (float)Q16[(long)b * DM_ + j];
  __syncthreads();
  if (tid < H_ * N_) {
    int h = tid / N_, n = tid - h * N_;
    const f16* krow = K16 + ((long)b * N_ + n) * DM_ + h * DK_;
    float s = 0.f;
    for (int d = 0; d < DK_; ++d) s += sQ[h * DK_ + d] * (float)krow[d];
    sS[h][n] = s * SCALE_;
  }
  __syncthreads();
  if (tid < H_) {
    int h = tid;
    float vals[N_], mx = -3.4e38f;
    for (int n = 0; n < N_; ++n) {
      float v = mask[b * N_ + n] ? -1e10f : sS[h][n];
      vals[n] = v; mx = fmaxf(mx, v);
    }
    float den = 0.f;
    for (int n = 0; n < N_; ++n) { float e = __expf(vals[n] - mx); vals[n] = e; den += e; }
    float inv = 1.0f / den;
    for (int n = 0; n < N_; ++n) {
      float p = vals[n] * inv;
      sP[h][n] = p;
      attn_sq[((long)h * B_ + b) * N_ + n] = p;
    }
  }
  __syncthreads();
  for (int idx = tid; idx < H_ * DM_; idx += 128) {
    int h = idx / DM_, j = idx - h * DM_;
    float s = 0.f;
    for (int n = 0; n < N_; ++n) s += sP[h][n] * (float)kin16[((long)b * N_ + n) * DM_ + j];
    pk16[(long)b * (H_ * DM_) + idx] = (f16)s;
  }
}

// ---------------- kernel 4: attn_out = pk@Wv_h^T ; out = attn_out@fc_w^T + b + q_in ; LN ----
__global__ __launch_bounds__(256) void aofc_ln_kernel(const f16* __restrict__ pk16,
                                                      const f16* __restrict__ Wv16,
                                                      const f16* __restrict__ fcw16,
                                                      const float* __restrict__ fc_b,
                                                      const float* __restrict__ qinf,
                                                      const float* __restrict__ ln_g,
                                                      const float* __restrict__ ln_b,
                                                      f16* __restrict__ aoln16) {
  __shared__ __align__(16) f16 ao[64 * 384];
  __shared__ float sSum[64], sSq[64], sMu[64], sRs[64];
  const int  tid  = threadIdx.x;
  const long row0 = (long)blockIdx.x * 64;
  if (tid < 64) { sSum[tid] = 0.f; sSq[tid] = 0.f; }
  const int w = tid >> 5, lane = tid & 31;
  const int mg = w >> 2, ng = w & 3;       // ng also selects the head (96 cols/head)
  const int r = lane & 15, half = lane >> 4;

  { // phase 1: head-projected attention output into LDS (f16)
    const f16* a0 = pk16 + (row0 + mg * 32 + r) * (H_ * DM_) + ng * DM_;
    const f16* a1 = a0 + (long)16 * (H_ * DM_);
    const f16* brow[6];
#pragma unroll
    for (int nt = 0; nt < 6; ++nt) brow[nt] = Wv16 + (ng * 96 + nt * 16 + r) * DM_;
    v8f acc[2][6] = {};
    for (int k0 = 0; k0 < DM_; k0 += 32) {
      v16h af0 = ldA(a0, k0, half);
      v16h af1 = ldA(a1, k0, half);
#pragma unroll
      for (int nt = 0; nt < 6; ++nt) {
        v16h bf = ldB(brow[nt], k0, half);
        acc[0][nt] = wmma(af0, bf, acc[0][nt]);
        acc[1][nt] = wmma(af1, bf, acc[1][nt]);
      }
    }
#pragma unroll
    for (int mt = 0; mt < 2; ++mt)
#pragma unroll
      for (int nt = 0; nt < 6; ++nt)
#pragma unroll
        for (int i = 0; i < 8; ++i)
          ao[(mg * 32 + mt * 16 + half * 8 + i) * DM_ + ng * 96 + nt * 16 + r] =
              (f16)acc[mt][nt][i];
  }
  __syncthreads();

  // phase 2: FC on LDS-resident activation
  v8f acc2[2][6] = {};
  {
    const f16* a0 = ao + (mg * 32 + r) * DM_;
    const f16* a1 = a0 + 16 * DM_;
    const f16* brow[6];
#pragma unroll
    for (int nt = 0; nt < 6; ++nt) brow[nt] = fcw16 + (ng * 96 + nt * 16 + r) * DM_;
    for (int k0 = 0; k0 < DM_; k0 += 32) {
      v16h af0 = ldA(a0, k0, half);
      v16h af1 = ldA(a1, k0, half);
#pragma unroll
      for (int nt = 0; nt < 6; ++nt) {
        v16h bf = ldB(brow[nt], k0, half);
        acc2[0][nt] = wmma(af0, bf, acc2[0][nt]);
        acc2[1][nt] = wmma(af1, bf, acc2[1][nt]);
      }
    }
  }
  // bias + residual; LN statistics via LDS float atomics (pre-reduced over the 6 cols/lane)
#pragma unroll
  for (int mt = 0; mt < 2; ++mt)
#pragma unroll
    for (int i = 0; i < 8; ++i) {
      int   rl = mg * 32 + mt * 16 + half * 8 + i;
      float ls = 0.f, lq = 0.f;
#pragma unroll
      for (int nt = 0; nt < 6; ++nt) {
        int   col = ng * 96 + nt * 16 + r;
        float v   = acc2[mt][nt][i] + fc_b[col] + qinf[(row0 + rl) * DM_ + col];
        acc2[mt][nt][i] = v;
        ls += v; lq += v * v;
      }
      atomicAdd(&sSum[rl], ls);
      atomicAdd(&sSq[rl], lq);
    }
  __syncthreads();
  if (tid < 64) {
    float mu  = sSum[tid] * (1.0f / DM_);
    float var = sSq[tid] * (1.0f / DM_) - mu * mu;
    sMu[tid] = mu;
    sRs[tid] = rsqrtf(var + 1e-5f);
  }
  __syncthreads();
#pragma unroll
  for (int mt = 0; mt < 2; ++mt)
#pragma unroll
    for (int nt = 0; nt < 6; ++nt)
#pragma unroll
      for (int i = 0; i < 8; ++i) {
        int   rl  = mg * 32 + mt * 16 + half * 8 + i;
        int   col = ng * 96 + nt * 16 + r;
        float o   = (acc2[mt][nt][i] - sMu[rl]) * sRs[rl] * ln_g[col] + ln_b[col];
        aoln16[(row0 + rl) * DM_ + col] = (f16)o;
      }
}

// ---------------- kernel 5: MLP: x=[out_ln, src]; y = relu(x@m1^T+b1)@m2^T + b2 ----------------
// 32 rows per block; wave tile 16 x 32
__global__ __launch_bounds__(256) void mlp_kernel(const f16* __restrict__ aoln16,
                                                  const f16* __restrict__ src16,
                                                  const f16* __restrict__ m1w16,
                                                  const float* __restrict__ m1_b,
                                                  const f16* __restrict__ m2w16,
                                                  const float* __restrict__ m2_b,
                                                  float* __restrict__ y) {
  __shared__ __align__(16) f16 xs[32 * 512];
  __shared__ __align__(16) f16 hs[32 * 128];
  const int  tid  = threadIdx.x;
  const long row0 = (long)blockIdx.x * 32;
  for (int idx = tid; idx < 32 * 512; idx += 256) {
    int rl = idx >> 9, c = idx & 511;
    xs[idx] = (c < DM_) ? aoln16[(row0 + rl) * DM_ + c]
                        : src16[(row0 + rl) * D_ + (c - DM_)];
  }
  __syncthreads();
  const int w = tid >> 5, lane = tid & 31;
  const int mg = w >> 2, ng = w & 3;
  const int r = lane & 15, half = lane >> 4;

  { // GEMM1: (32 x 512) @ (128 x 512)^T, ReLU + bias -> hs
    const f16* a0 = xs + (mg * 16 + r) * 512;
    const f16* b0 = m1w16 + (ng * 32 + r) * 512;
    const f16* b1 = m1w16 + (ng * 32 + 16 + r) * 512;
    v8f acc[2] = {};
    for (int k0 = 0; k0 < 512; k0 += 32) {
      v16h af = ldA(a0, k0, half);
      acc[0] = wmma(af, ldB(b0, k0, half), acc[0]);
      acc[1] = wmma(af, ldB(b1, k0, half), acc[1]);
    }
#pragma unroll
    for (int nt = 0; nt < 2; ++nt)
#pragma unroll
      for (int i = 0; i < 8; ++i) {
        int   rl  = mg * 16 + half * 8 + i;
        int   col = ng * 32 + nt * 16 + r;
        float v   = fmaxf(acc[nt][i] + m1_b[col], 0.0f);
        hs[rl * 128 + col] = (f16)v;
      }
  }
  __syncthreads();
  { // GEMM2: (32 x 128) @ (128 x 128)^T -> y (f32 output)
    const f16* a0 = hs + (mg * 16 + r) * 128;
    const f16* b0 = m2w16 + (ng * 32 + r) * 128;
    const f16* b1 = m2w16 + (ng * 32 + 16 + r) * 128;
    v8f acc[2] = {};
    for (int k0 = 0; k0 < 128; k0 += 32) {
      v16h af = ldA(a0, k0, half);
      acc[0] = wmma(af, ldB(b0, k0, half), acc[0]);
      acc[1] = wmma(af, ldB(b1, k0, half), acc[1]);
    }
#pragma unroll
    for (int nt = 0; nt < 2; ++nt)
#pragma unroll
      for (int i = 0; i < 8; ++i) {
        int rl  = mg * 16 + half * 8 + i;
        int col = ng * 32 + nt * 16 + r;
        y[(row0 + rl) * D_ + col] = acc[nt][i] + m2_b[col];
      }
  }
}

// ---------------- host side ----------------
extern "C" void kernel_launch(void* const* d_in, const int* in_sizes, int n_in,
                              void* d_out, int out_size, void* d_ws, size_t ws_size,
                              hipStream_t stream) {
  const float* src     = (const float*)d_in[0];
  const float* seq     = (const float*)d_in[1];
  const float* seq_e   = (const float*)d_in[2];
  const float* src_ts  = (const float*)d_in[3];
  const float* seq_ts  = (const float*)d_in[4];
  const unsigned char* mask = (const unsigned char*)d_in[5];
  const float* bf      = (const float*)d_in[6];
  const float* ph      = (const float*)d_in[7];
  const float* Wq      = (const float*)d_in[8];
  const float* Wk      = (const float*)d_in[9];
  const float* Wv      = (const float*)d_in[10];
  const float* fc_w    = (const float*)d_in[11];
  const float* fc_b    = (const float*)d_in[12];
  const float* ln_g    = (const float*)d_in[13];
  const float* ln_b    = (const float*)d_in[14];
  const float* m1_w    = (const float*)d_in[15];
  const float* m1_b    = (const float*)d_in[16];
  const float* m2_w    = (const float*)d_in[17];
  const float* m2_b    = (const float*)d_in[18];

  float* y       = (float*)d_out;                       // (B, 128)
  float* attn_sq = (float*)d_out + (long)B_ * D_;       // (H*B, N)

  // workspace carve (bytes, 256-aligned)
  char* p = (char*)d_ws;
  auto carve = [&](size_t bytes) { char* r = p; p += (bytes + 255) & ~(size_t)255; return r; };
  f16*   kin16  = (f16*)  carve(BN_ * DM_ * sizeof(f16));          // ~252 MB
  f16*   qin16  = (f16*)  carve((long)B_ * DM_ * sizeof(f16));
  float* qinf   = (float*)carve((long)B_ * DM_ * sizeof(float));
  f16*   src16  = (f16*)  carve((long)B_ * D_ * sizeof(f16));
  f16*   Q16    = (f16*)  carve((long)B_ * DM_ * sizeof(f16));
  f16*   K16    = (f16*)  carve(BN_ * DM_ * sizeof(f16));          // ~252 MB
  f16*   pk16   = (f16*)  carve((long)B_ * H_ * DM_ * sizeof(f16));
  f16*   aoln16 = (f16*)  carve((long)B_ * DM_ * sizeof(f16));
  f16*   Wq16   = (f16*)  carve(DM_ * DM_ * sizeof(f16));
  f16*   Wk16   = (f16*)  carve(DM_ * DM_ * sizeof(f16));
  f16*   Wv16   = (f16*)  carve(DM_ * DM_ * sizeof(f16));
  f16*   fcw16  = (f16*)  carve(DM_ * DM_ * sizeof(f16));
  f16*   m1w16  = (f16*)  carve(D_ * (DM_ + D_) * sizeof(f16));
  f16*   m2w16  = (f16*)  carve(D_ * D_ * sizeof(f16));

  auto cvt = [&](const float* in, f16* out, int n) {
    cvt_f32_f16<<<(n + 255) / 256, 256, 0, stream>>>(in, out, n);
  };
  cvt(Wq, Wq16, DM_ * DM_);
  cvt(Wk, Wk16, DM_ * DM_);
  cvt(Wv, Wv16, DM_ * DM_);
  cvt(fc_w, fcw16, DM_ * DM_);
  cvt(m1_w, m1w16, D_ * (DM_ + D_));
  cvt(m2_w, m2w16, D_ * D_);

  pack_kin<<<(int)((BN_ * DM_) / 256), 256, 0, stream>>>(seq, seq_e, seq_ts, bf, ph, kin16);
  pack_qin<<<(int)(((long)B_ * DM_) / 256), 256, 0, stream>>>(src, src_ts, bf, ph, qin16, qinf);
  pack_src16<<<(int)(((long)B_ * D_) / 256), 256, 0, stream>>>(src, src16);

  gemm_kn384<<<(int)(BN_ / 64), 256, 0, stream>>>(kin16, Wk16, K16);     // K projection
  gemm_kn384<<<B_ / 64, 256, 0, stream>>>(qin16, Wq16, Q16);             // Q projection

  attn_kernel<<<B_, 128, 0, stream>>>(Q16, K16, kin16, mask, attn_sq, pk16);

  aofc_ln_kernel<<<B_ / 64, 256, 0, stream>>>(pk16, Wv16, fcw16, fc_b, qinf, ln_g, ln_b, aoln16);

  mlp_kernel<<<B_ / 32, 256, 0, stream>>>(aoln16, src16, m1w16, m1_b, m2w16, m2_b, y);
}
